// MNDRE_10428180594965
// MI455X (gfx1250) — compile-verified, matrix operand order checked by reading, not run on hardware
//
#include <hip/hip_runtime.h>

typedef __attribute__((ext_vector_type(16))) __bf16 v16bf;
typedef __attribute__((ext_vector_type(8)))  float  v8f;

#define DD      8
#define HH      64
#define WIN     16
#define NOUT    4
#define WIDTH   128
#define IHID    64
#define NPAIR   28
#define LOGSIG  36
#define NW      32
#define LPTS    513
#define DT      0.03125f   /* WIN/(L-1) = 16/512 */

static __device__ const int IU_I[NPAIR] =
  {0,0,0,0,0,0,0, 1,1,1,1,1,1, 2,2,2,2,2, 3,3,3,3, 4,4,4, 5,5, 6};
static __device__ const int IU_J[NPAIR] =
  {1,2,3,4,5,6,7, 2,3,4,5,6,7, 3,4,5,6,7, 4,5,6,7, 5,6,7, 6,7, 7};

__device__ __forceinline__ unsigned short f2bf(float f) {
  unsigned int u = __float_as_uint(f);
  unsigned int r = u + 0x7FFFu + ((u >> 16) & 1u);   // round-to-nearest-even
  return (unsigned short)(r >> 16);
}

union Frag { v16bf v; unsigned int u[8]; };

// uint-offset of WMMA bf16 operand pair v inside one K-tile (32 K values):
// ISA 16-bit A/B layout: VGPR v<4 -> K=2v, v>=4 -> K=16+2(v-4); hi lanes +8.
__device__ __forceinline__ int koff(int v, int hi4, int kbase16) {
  return ((v < 4) ? v : (v + 4)) + hi4 + kbase16;
}

// ---------------------------------------------------------------------------
// vf: k = A(z) @ g  where A = [f | Lie brackets [f_i,f_j]]
// Jacobian never materialized:  G[i,ja] = tanh'[ja] * (W3 @ (M2 @ f^T))[ja,i]
// ---------------------------------------------------------------------------
__device__ __forceinline__ void vf(
    const float* __restrict__ z,
    const float* __restrict__ W1, const float* __restrict__ b1,
    const float* __restrict__ W2, const float* __restrict__ b2,
    const float* __restrict__ W3, const float* __restrict__ b3,
    const unsigned int* __restrict__ w2bf, const unsigned int* __restrict__ w3bf,
    unsigned short* sM1T, unsigned short* sQT, unsigned short* sM2h,
    unsigned short* sFbf,
    float* sG, float* sF, float* sH1, float* sH2, float* sD2,
    const float* sGv, float* sK, int tid)
{
  const int lane = tid & 31;
  const int wave = tid >> 5;
  const int lrow = lane & 15;
  const int hi4  = (lane >= 16) ? 4 : 0;

  // ---- layer 1: h1 = lipswish(W1 z + b1); M1T[c][k] = d1[k]*W1[k][c] (bf16)
  if (tid < WIDTH) {
    const float* wrow = W1 + tid * HH;
    float p = b1[tid];
    for (int j = 0; j < HH; ++j) p += wrow[j] * z[j];
    float s = 1.f / (1.f + __expf(-p));
    sH1[tid] = 0.909f * p * s;
    float d1 = 0.909f * s * (1.f + p * (1.f - s));
    for (int c = 0; c < HH; ++c) sM1T[c * WIDTH + tid] = f2bf(d1 * wrow[c]);
  }
  __syncthreads();

  // ---- layer 2: h2, d2
  if (tid < WIDTH) {
    const float* wrow = W2 + tid * WIDTH;
    float p = b2[tid];
    for (int j = 0; j < WIDTH; ++j) p += wrow[j] * sH1[j];
    float s = 1.f / (1.f + __expf(-p));
    sH2[tid] = 0.909f * p * s;
    sD2[tid] = 0.909f * s * (1.f + p * (1.f - s));
  }
  __syncthreads();

  // ---- layer 3: f = tanh(W3 h2 + b3), full f32 (level-1 signal path);
  //      bf16 copy feeds only the bracket-correction GEMMs
  for (int r = tid; r < DD * HH; r += 256) {
    const float* wrow = W3 + r * WIDTH;
    float p = b3[r];
    for (int j = 0; j < WIDTH; ++j) p += wrow[j] * sH2[j];
    const float fv = tanhf(p);
    sF[r]   = fv;
    sFbf[r] = f2bf(fv);
  }
  __syncthreads();

  // ---- GEMM 1 (WMMA): M2 = diag(d2) * (W2 @ M1)   [128x128 @ 128x64]
  {
    const unsigned int* m1u = (const unsigned int*)sM1T;
    const int mt = wave;                     // 8 waves -> 8 row tiles
    const int m0 = mt * 16 + lrow;
    for (int nt = 0; nt < 4; ++nt) {
      v8f acc = {};
      const int n = nt * 16 + lrow;
      for (int kt = 0; kt < 4; ++kt) {
        Frag a, bb;
        const int kb = kt * 16;
#pragma unroll
        for (int v = 0; v < 8; ++v) {
          const int ko = koff(v, hi4, kb);
          a.u[v]  = w2bf[m0 * 64 + ko];
          bb.u[v] = m1u[n * 64 + ko];
        }
        acc = __builtin_amdgcn_wmma_f32_16x16x32_bf16(
                  false, a.v, false, bb.v, (short)0, acc, false, false);
      }
#pragma unroll
      for (int r = 0; r < 8; ++r) {
        const int row = mt * 16 + ((hi4 ? 8 : 0) + r);
        const int col = nt * 16 + lrow;
        sM2h[row * HH + col] = f2bf(acc[r] * sD2[row]);
      }
    }
  }
  __syncthreads();

  // ---- GEMM Q (WMMA): Q = M2 @ f^T   [128x64 @ 64x16, cols 8..15 zero]
  //      C-fragment written transposed: lane's column index == QT row.
  {
    const unsigned int* m2u = (const unsigned int*)sM2h;   // 128 x 64 bf16
    const unsigned int* fbu = (const unsigned int*)sFbf;   // 16 x 64 bf16 (N x K)
    const int mt = wave;                     // 8 row tiles -> M=128
    const int m0 = mt * 16 + lrow;
    v8f acc = {};
    for (int kt = 0; kt < 2; ++kt) {         // K = 64
      Frag a, bb;
      const int kb = kt * 16;
#pragma unroll
      for (int v = 0; v < 8; ++v) {
        const int ko = koff(v, hi4, kb);
        a.u[v]  = m2u[m0 * 32 + ko];
        bb.u[v] = fbu[lrow * 32 + ko];
      }
      acc = __builtin_amdgcn_wmma_f32_16x16x32_bf16(
                false, a.v, false, bb.v, (short)0, acc, false, false);
    }
#pragma unroll
    for (int r = 0; r < 8; ++r) {
      const int row = mt * 16 + ((hi4 ? 8 : 0) + r);       // Q row (c index)
      sQT[lrow * WIDTH + row] = f2bf(acc[r]);              // QT[i][c]
    }
  }
  __syncthreads();

  // ---- GEMM 2 (WMMA): R = W3 @ Q ; G[i][ja] = (1-f^2)[ja] * R[ja][i]
  {
    const unsigned int* qtu = (const unsigned int*)sQT;    // rows 8..15 zero
    for (int mq = 0; mq < 4; ++mq) {
      const int mt = wave + mq * 8;          // 32 row tiles of 16 -> M=512
      const int m0 = mt * 16 + lrow;
      v8f acc = {};
      for (int kt = 0; kt < 4; ++kt) {
        Frag a, bb;
        const int kb = kt * 16;
#pragma unroll
        for (int v = 0; v < 8; ++v) {
          const int ko = koff(v, hi4, kb);
          a.u[v]  = w3bf[m0 * 64 + ko];
          bb.u[v] = qtu[lrow * 64 + ko];     // N tile 0 (cols 0..15)
        }
        acc = __builtin_amdgcn_wmma_f32_16x16x32_bf16(
                  false, a.v, false, bb.v, (short)0, acc, false, false);
      }
      if (lrow < DD) {
#pragma unroll
        for (int r = 0; r < 8; ++r) {
          const int ja = mt * 16 + ((hi4 ? 8 : 0) + r);
          const float fv = sF[ja];
          sG[lrow * 512 + ja] = acc[r] * (1.f - fv * fv);
        }
      }
    }
  }
  __syncthreads();

  // ---- k[a] = sum_i f[i,a] g[i] + sum_p bracket_p[a] g[8+p]
  if (tid < HH) {
    float acc = 0.f;
#pragma unroll
    for (int i = 0; i < DD; ++i) acc += sF[i * HH + tid] * sGv[i];
#pragma unroll
    for (int p = 0; p < NPAIR; ++p) {
      const int i = IU_I[p], j = IU_J[p];
      acc += (sG[i * 512 + j * HH + tid] - sG[j * 512 + i * HH + tid]) * sGv[DD + p];
    }
    sK[tid] = acc;
  }
  __syncthreads();
}

// ---------------------------------------------------------------------------
// one-shot weight conversion to bf16 (L2-resident, shared across the grid)
// ---------------------------------------------------------------------------
__global__ void prep_bf16(const float* __restrict__ W2, const float* __restrict__ W3,
                          unsigned short* w2bf, unsigned short* w3bf) {
  const int i = blockIdx.x * blockDim.x + threadIdx.x;
  if (i < 128 * 128) w2bf[i] = f2bf(W2[i]);
  if (i < 512 * 128) w3bf[i] = f2bf(W3[i]);
}

// ---------------------------------------------------------------------------
// main: one workgroup (8 wave32) per batch element; full scan in-LDS
// ---------------------------------------------------------------------------
__global__ __launch_bounds__(256)
void cde_scan(const float* __restrict__ cv,
              const float* __restrict__ Wi1, const float* __restrict__ bi1,
              const float* __restrict__ Wi2, const float* __restrict__ bi2,
              const float* __restrict__ W1,  const float* __restrict__ b1,
              const float* __restrict__ W2,  const float* __restrict__ b2,
              const float* __restrict__ W3,  const float* __restrict__ b3,
              const float* __restrict__ Wr,  const float* __restrict__ br,
              const float* __restrict__ shf,
              const unsigned int* __restrict__ w2bf,
              const unsigned int* __restrict__ w3bf,
              float* __restrict__ out)
{
  __shared__ unsigned short sM1T[HH * WIDTH];      // 16 KB  (M1^T, 64x128 bf16)
  __shared__ unsigned short sQT[16 * WIDTH];       //  4 KB  (Q^T, 16x128 bf16)
  __shared__ unsigned short sM2h[WIDTH * HH];      // 16 KB  (M2, 128x64 bf16)
  __shared__ unsigned short sFbf[16 * HH];         //  2 KB  (f, 16x64 bf16, rows 8..15 = 0)
  __shared__ float sG[DD * 512];                   // 16 KB
  __shared__ float sF[512];                        //  2 KB
  __shared__ float sH1[WIDTH], sH2[WIDTH], sD2[WIDTH];
  __shared__ float sY[HH], sYt[HH], sK1[HH], sK[HH], sLip[IHID];
  __shared__ float sGv[LOGSIG];
  __shared__ float sLogs[NW * LOGSIG];             // 4.5 KB

  const int b   = blockIdx.x;
  const int tid = threadIdx.x;
  const float* v = cv + b * LPTS * DD;

  for (int i = tid; i < 16 * WIDTH; i += 256) sQT[i] = 0;   // pad rows 8..15
  for (int i = tid; i < 16 * HH;    i += 256) sFbf[i] = 0;  // pad rows 8..15

  // ---- phase A: depth-2 Lyndon log-signatures per window
  for (int idx = tid; idx < NW * LOGSIG; idx += 256) {
    const int w = idx / LOGSIG, c = idx % LOGSIG;
    const float* vb = v + w * WIN * DD;
    float r;
    if (c < DD) {
      r = vb[WIN * DD + c] - vb[c];                 // telescoped increment sum
    } else {
      const int p = c - DD, i = IU_I[p], j = IU_J[p];
      float Pi = 0.f, Pj = 0.f, acc = 0.f;
      for (int t = 0; t < WIN; ++t) {
        const float ii = vb[(t + 1) * DD + i] - vb[t * DD + i];
        const float jj = vb[(t + 1) * DD + j] - vb[t * DD + j];
        acc += Pi * jj - Pj * ii;                   // Levy area
        Pi += ii; Pj += jj;
      }
      r = 0.5f * acc;
    }
    sLogs[idx] = r;
  }
  __syncthreads();

  // ---- phase B: y0 = Wi2 lipswish(Wi1 x0 + bi1) + bi2
  if (tid < IHID) {
    float p = bi1[tid];
    for (int d = 0; d < DD; ++d) p += Wi1[tid * DD + d] * v[d];
    const float s = 1.f / (1.f + __expf(-p));
    sLip[tid] = 0.909f * p * s;
  }
  __syncthreads();
  if (tid < HH) {
    float p = bi2[tid];
    for (int k = 0; k < IHID; ++k) p += Wi2[tid * IHID + k] * sLip[k];
    sY[tid] = p;
  }
  __syncthreads();

  const float shift = shf[0];
  if (tid < NOUT) {
    float p = br[tid] + shift;
    for (int a = 0; a < HH; ++a) p += Wr[tid * HH + a] * sY[a];
    out[b * (NW + 1) * NOUT + tid] = p;
  }
  __syncthreads();

  // ---- phase C: Heun scan over windows
  for (int w = 0; w < NW; ++w) {
    if (tid < LOGSIG) sGv[tid] = sLogs[w * LOGSIG + tid] * (1.0f / DT);
    __syncthreads();

    vf(sY, W1, b1, W2, b2, W3, b3, w2bf, w3bf,
       sM1T, sQT, sM2h, sFbf, sG, sF, sH1, sH2, sD2, sGv, sK, tid);  // k1

    if (tid < HH) { sK1[tid] = sK[tid]; sYt[tid] = sY[tid] + DT * sK[tid]; }
    __syncthreads();

    vf(sYt, W1, b1, W2, b2, W3, b3, w2bf, w3bf,
       sM1T, sQT, sM2h, sFbf, sG, sF, sH1, sH2, sD2, sGv, sK, tid);  // k2

    if (tid < HH) sY[tid] += 0.5f * DT * (sK1[tid] + sK[tid]);
    __syncthreads();

    if (tid < NOUT) {
      float p = br[tid] + shift;
      for (int a = 0; a < HH; ++a) p += Wr[tid * HH + a] * sY[a];
      out[b * (NW + 1) * NOUT + (w + 1) * NOUT + tid] = p;
    }
    __syncthreads();
  }
}

extern "C" void kernel_launch(void* const* d_in, const int* in_sizes, int n_in,
                              void* d_out, int out_size, void* d_ws, size_t ws_size,
                              hipStream_t stream) {
  const float* cv  = (const float*)d_in[0];
  const float* Wi1 = (const float*)d_in[1];
  const float* bi1 = (const float*)d_in[2];
  const float* Wi2 = (const float*)d_in[3];
  const float* bi2 = (const float*)d_in[4];
  const float* W1  = (const float*)d_in[5];
  const float* b1  = (const float*)d_in[6];
  const float* W2  = (const float*)d_in[7];
  const float* b2  = (const float*)d_in[8];
  const float* W3  = (const float*)d_in[9];
  const float* b3  = (const float*)d_in[10];
  const float* Wr  = (const float*)d_in[11];
  const float* br  = (const float*)d_in[12];
  const float* shf = (const float*)d_in[13];

  unsigned short* w2bf = (unsigned short*)d_ws;                          // 32 KB
  unsigned short* w3bf = (unsigned short*)((char*)d_ws + 128 * 128 * 2); // 128 KB

  prep_bf16<<<256, 256, 0, stream>>>(W2, W3, w2bf, w3bf);
  cde_scan<<<32, 256, 0, stream>>>(cv, Wi1, bi1, Wi2, bi2, W1, b1, W2, b2,
                                   W3, b3, Wr, br, shf,
                                   (const unsigned int*)w2bf,
                                   (const unsigned int*)w3bf,
                                   (float*)d_out);
}
